// DecoderRNN_36369783062974
// MI455X (gfx1250) — matrix operand order repaired and checked by reading, not compiled, not run on hardware
//
#include <hip/hip_runtime.h>
#include <stdint.h>

// ---------------------------------------------------------------------------
// DecoderRNN (LSTM, eval path) for MI455X / gfx1250, wave32, bf16 WMMA + TDM.
//   V=32000, E=512, H=1024, CTX=1024, T=64, B=64
// Strategy:
//   * Hoist context-part / embedding-part of the gates out of the scan.
//   * Defer the vocab projection (72% of FLOPs) to one big parallel GEMM.
//   * bf16 operands + f32 accumulation via V_WMMA_F32_16X16X32_BF16.
//   * W panels staged to LDS by the Tensor Data Mover, double-buffered and
//     overlapped with WMMA via s_wait_tensorcnt (weights are L2-resident).
// ---------------------------------------------------------------------------

typedef __attribute__((ext_vector_type(16))) __bf16 v16bf;
typedef __attribute__((ext_vector_type(8)))  float  v8f;
typedef __attribute__((ext_vector_type(4)))  unsigned int u32x4;
typedef __attribute__((ext_vector_type(8)))  int i32x8;
typedef __attribute__((ext_vector_type(4)))  int i32x4;

#define VOC 32000
#define EMB 512
#define HID 1024
#define CTXW 1024
#define TT 64
#define BB 64
#define G4H 4096   /* 4*HID */
#define XW 1536    /* CTX + E */

#if __has_builtin(__builtin_amdgcn_tensor_load_to_lds)
#define USE_TDM 1
#else
#define USE_TDM 0
#endif

__device__ __forceinline__ unsigned short f2bf(float f) {
    union { float f; uint32_t u; } v; v.f = f;
    uint32_t r = v.u + 0x7FFFu + ((v.u >> 16) & 1u);   // round-to-nearest-even
    return (unsigned short)(r >> 16);
}
__device__ __forceinline__ float sigmf(float x) {
    return 1.0f / (1.0f + __expf(-x));
}

__device__ __forceinline__ void wait_tensorcnt_le1() {
#if __has_builtin(__builtin_amdgcn_s_wait_tensorcnt)
    __builtin_amdgcn_s_wait_tensorcnt(1);
#else
    asm volatile("s_wait_tensorcnt 0x1" ::: "memory");
#endif
}
__device__ __forceinline__ void wait_tensorcnt_le0() {
#if __has_builtin(__builtin_amdgcn_s_wait_tensorcnt)
    __builtin_amdgcn_s_wait_tensorcnt(0);
#else
    asm volatile("s_wait_tensorcnt 0x0" ::: "memory");
#endif
}

#if USE_TDM
// TDM: DMA a 64x64 bf16 tile (rows n, 64 contiguous K each) of row-major
// W[N,K] (row stride ldw elements) into LDS at byte offset lds_off.
// D# layout per CDNA5 ISA ch.8: group0 = {count/flags, lds_addr, gaddr};
// group1 = {data_size=2B, tensor_dim0/1 = tile dims (fully in-bounds),
// tile_dim0/1 = 64, tensor_dim0_stride = ldw}; groups 2/3 zero (2D tensor).
__device__ __forceinline__ void tdm_load_64x64(const unsigned short* gptr,
                                               unsigned int lds_off, int ldw) {
    unsigned long long ga = (unsigned long long)gptr;
    u32x4 g0 = { 1u,                                     // count=1, user mode
                 lds_off,                                // lds_addr (bytes)
                 (unsigned int)ga,                       // global_addr[31:0]
                 (unsigned int)((ga >> 32) & 0x1FFFFFFu) // global_addr[56:32]
                 | (2u << 30) };                         // type = 2 ("image")
    i32x8 g1 = { (int)(1u << 16),          // data_size = 1 (2 bytes)
                 (int)(64u << 16),         // tensor_dim0[15:0] = 64  @bit48
                 (int)(64u << 16),         // dim0_hi=0 | tensor_dim1_lo = 64
                 (int)(64u << 16),         // dim1_hi=0 | tile_dim0 = 64
                 (int)64,                  // tile_dim1 = 64, tile_dim2 = 0
                 (int)(unsigned)ldw,       // tensor_dim0_stride[31:0]
                 0, 0 };                   // stride_hi, dim1_stride = 0
    i32x4 z4 = { 0, 0, 0, 0 };
#if __clang_major__ >= 23
    i32x8 z8 = { 0, 0, 0, 0, 0, 0, 0, 0 };
    __builtin_amdgcn_tensor_load_to_lds(g0, g1, z4, z4, z8, 0);
#else
    __builtin_amdgcn_tensor_load_to_lds(g0, g1, z4, z4, 0);
#endif
}
#endif

// --------------------------- elementwise kernels ---------------------------

__global__ void k_f32_to_bf16(const float* __restrict__ src,
                              unsigned short* __restrict__ dst, int n) {
    int i = blockIdx.x * blockDim.x + threadIdx.x;
    if (i < n) dst[i] = f2bf(src[i]);
}

__global__ void k_bias_sum(const float* __restrict__ a, const float* __restrict__ b,
                           float* __restrict__ o, int n) {
    int i = blockIdx.x * blockDim.x + threadIdx.x;
    if (i < n) o[i] = a[i] + b[i];
}

__global__ void k_gather_emb(const int* __restrict__ seq,
                             const float* __restrict__ emb,
                             unsigned short* __restrict__ xe) {
    int i = blockIdx.x * blockDim.x + threadIdx.x;      // T*B*E threads
    int tb = i >> 9;                                    // /E (E=512)
    int e  = i & (EMB - 1);
    int tok = seq[tb];
    xe[i] = f2bf(emb[(size_t)tok * EMB + e]);
}

__global__ void k_lstm_cell(const float* __restrict__ gates,
                            float* __restrict__ cst,
                            unsigned short* __restrict__ hc) {
    int i = blockIdx.x * blockDim.x + threadIdx.x;      // B*H threads
    int b = i >> 10;                                    // /H
    int j = i & (HID - 1);
    const float* g = gates + (size_t)b * G4H;
    float ig = sigmf(g[j]);
    float fg = sigmf(g[HID + j]);
    float gg = tanhf(g[2 * HID + j]);
    float og = sigmf(g[3 * HID + j]);
    float cn = fg * cst[i] + ig * gg;
    float hn = og * tanhf(cn);
    cst[i] = cn;
    hc[(size_t)b * (2 * HID) + j]       = f2bf(hn);
    hc[(size_t)b * (2 * HID) + HID + j] = f2bf(cn);
}

// ------------------------------ GEMM common --------------------------------
// Y[M,N] = act( A[M,K](bf16) @ W[N,K].T(bf16) + bias[N] + addsrc[row%mod, N] )
// Block = 128 threads = 4 waves; block tile 64x64; wave tile 16x64.
// All call sites: M mult of 16 per wave, N mult of 64, K mult of 64.

__device__ __forceinline__ void gemm_epilogue(
    const v8f* acc, int m0, int n0, int half, int r16,
    const float* bias, const float* addsrc, int add_ld, int add_mod,
    float* outF, int ldoF, unsigned short* outH, int ldoH, int act) {
#pragma unroll
    for (int tn = 0; tn < 4; ++tn) {
        const int n = n0 + tn * 16 + r16;
        const float bv = bias ? bias[n] : 0.0f;
#pragma unroll
        for (int r = 0; r < 8; ++r) {
            const int row = m0 + r + 8 * half;
            float v = acc[tn][r] + bv;
            if (addsrc) v += addsrc[(size_t)(row % add_mod) * add_ld + n];
            if (act) v = tanhf(v);
            if (outF) outF[(size_t)row * ldoF + n] = v;
            if (outH) outH[(size_t)row * ldoH + n] = f2bf(v);
        }
    }
}

// Direct-load GEMM (no LDS): used for the small setup GEMMs.
__global__ __launch_bounds__(128)
void k_gemm_bf16(const unsigned short* __restrict__ A, int lda,
                 const unsigned short* __restrict__ W, int ldw,
                 const float* __restrict__ bias,
                 const float* __restrict__ addsrc, int add_ld, int add_mod,
                 float* __restrict__ outF, int ldoF,
                 unsigned short* __restrict__ outH, int ldoH,
                 int M, int N, int K, int act) {
    const int wave = threadIdx.x >> 5;
    const int lid  = threadIdx.x & 31;
    const int half = lid >> 4;
    const int r16  = lid & 15;
    const int m0 = blockIdx.y * 64 + wave * 16;
    const int n0 = blockIdx.x * 64;

    v8f acc[4] = {};
    const unsigned short* arow = A + (size_t)(m0 + r16) * lda + 8 * half;

    for (int k0 = 0; k0 < K; k0 += 32) {
        union { uint4 q[2]; v16bf v; } au;
        au.q[0] = *(const uint4*)(arow + k0);        // K = 8*half + 0..7
        au.q[1] = *(const uint4*)(arow + k0 + 16);   // K = 16 + 8*half + 0..7
        v16bf bfrag[4];
#pragma unroll
        for (int tn = 0; tn < 4; ++tn) {             // batch all B loads
            const unsigned short* wrow =
                W + (size_t)(n0 + tn * 16 + r16) * ldw + k0 + 8 * half;
            union { uint4 q[2]; v16bf v; } bu;
            bu.q[0] = *(const uint4*)(wrow);
            bu.q[1] = *(const uint4*)(wrow + 16);
            bfrag[tn] = bu.v;
        }
#pragma unroll
        for (int tn = 0; tn < 4; ++tn)
            acc[tn] = __builtin_amdgcn_wmma_f32_16x16x32_bf16(
                false, au.v, false, bfrag[tn], (short)0, acc[tn], false, false);
    }
    gemm_epilogue(acc, m0, n0, half, r16, bias, addsrc, add_ld, add_mod,
                  outF, ldoF, outH, ldoH, act);
}

// TDM-staged GEMM: W chunks (64 rows x 64 K, 8KB bf16) are DMA'd to LDS by
// the Tensor Data Mover, double-buffered, overlapped with WMMA.
__global__ __launch_bounds__(128)
void k_gemm_tdm(const unsigned short* __restrict__ A, int lda,
                const unsigned short* __restrict__ W, int ldw,
                const float* __restrict__ bias,
                const float* __restrict__ addsrc, int add_ld, int add_mod,
                float* __restrict__ outF, int ldoF,
                unsigned short* __restrict__ outH, int ldoH,
                int M, int N, int K, int act) {
    __shared__ __align__(16) unsigned short ldsB[2][64 * 64];   // 2 x 8KB
    const int wave = threadIdx.x >> 5;
    const int lid  = threadIdx.x & 31;
    const int half = lid >> 4;
    const int r16  = lid & 15;
    const int m0 = blockIdx.y * 64 + wave * 16;
    const int n0 = blockIdx.x * 64;
    const int nc = K >> 6;                            // 64-wide K chunks

    v8f acc[4] = {};
    const unsigned short* arow  = A + (size_t)(m0 + r16) * lda + 8 * half;
    const unsigned short* wpanel = W + (size_t)n0 * ldw;

#if USE_TDM
    // wave0 drives the TDM (scalar branch so other waves skip the issue)
    const bool issuer = (__builtin_amdgcn_readfirstlane((int)threadIdx.x) < 32);
    const unsigned int lds_base =
        (unsigned int)(unsigned long long)&ldsB[0][0];  // flat[31:0] = LDS off
    if (issuer) tdm_load_64x64(wpanel, lds_base, ldw);  // prefetch chunk 0
#endif

    for (int c = 0; c < nc; ++c) {
#if USE_TDM
        __syncthreads();   // everyone done reading buf[(c+1)&1] (iter c-1)
        if (issuer) {
            if (c + 1 < nc) {
                tdm_load_64x64(wpanel + (size_t)(c + 1) * 64,
                               lds_base + (unsigned)(((c + 1) & 1) * 8192),
                               ldw);
                wait_tensorcnt_le1();   // chunk c landed (in-order)
            } else {
                wait_tensorcnt_le0();
            }
        }
        __syncthreads();   // publish chunk c to all waves
#else
        __syncthreads();
        {   // cooperative staging fallback: 128 threads x 32 bf16
            int row  = threadIdx.x >> 1;
            int col0 = (threadIdx.x & 1) * 32;
            const unsigned short* src = wpanel + (size_t)row * ldw + c * 64 + col0;
            unsigned short* dst = &ldsB[c & 1][row * 64 + col0];
            ((uint4*)dst)[0] = ((const uint4*)src)[0];
            ((uint4*)dst)[1] = ((const uint4*)src)[1];
            ((uint4*)dst)[2] = ((const uint4*)src)[2];
            ((uint4*)dst)[3] = ((const uint4*)src)[3];
        }
        __syncthreads();
#endif
        const unsigned short* lbuf = &ldsB[c & 1][0];
        __builtin_prefetch(arow + (size_t)(c + 1) * 64, 0, 0);  // next A chunk
#pragma unroll
        for (int kk = 0; kk < 64; kk += 32) {
            union { uint4 q[2]; v16bf v; } au;
            const unsigned short* ap = arow + (size_t)c * 64 + kk;
            au.q[0] = *(const uint4*)(ap);
            au.q[1] = *(const uint4*)(ap + 16);
            v16bf bfrag[4];
#pragma unroll
            for (int tn = 0; tn < 4; ++tn) {          // B from LDS (ds_load)
                const unsigned short* lp =
                    lbuf + (tn * 16 + r16) * 64 + kk + 8 * half;
                union { uint4 q[2]; v16bf v; } bu;
                bu.q[0] = *(const uint4*)(lp);
                bu.q[1] = *(const uint4*)(lp + 16);
                bfrag[tn] = bu.v;
            }
#pragma unroll
            for (int tn = 0; tn < 4; ++tn)
                acc[tn] = __builtin_amdgcn_wmma_f32_16x16x32_bf16(
                    false, au.v, false, bfrag[tn], (short)0, acc[tn],
                    false, false);
        }
    }
    gemm_epilogue(acc, m0, n0, half, r16, bias, addsrc, add_ld, add_mod,
                  outF, ldoF, outH, ldoH, act);
}

// ------------------------------- launcher ----------------------------------

extern "C" void kernel_launch(void* const* d_in, const int* in_sizes, int n_in,
                              void* d_out, int out_size, void* d_ws, size_t ws_size,
                              hipStream_t stream) {
    (void)in_sizes; (void)n_in; (void)out_size; (void)ws_size;
    const int*   seq     = (const int*)  d_in[0];
    const float* context = (const float*)d_in[1];
    const float* emb     = (const float*)d_in[2];
    const float* W_ih    = (const float*)d_in[3];
    const float* b_ih    = (const float*)d_in[4];
    const float* W_hh    = (const float*)d_in[5];
    const float* b_hh    = (const float*)d_in[6];
    const float* W_initS = (const float*)d_in[7];
    const float* b_initS = (const float*)d_in[8];
    const float* W_initC = (const float*)d_in[9];
    const float* b_initC = (const float*)d_in[10];
    const float* W_d1    = (const float*)d_in[11];
    const float* b_d1    = (const float*)d_in[12];
    const float* W_d2    = (const float*)d_in[13];
    const float* b_d2    = (const float*)d_in[14];
    float* out = (float*)d_out;                     // [T,B,V]

    size_t off = 0;
    char* base = (char*)d_ws;
    auto carve = [&](size_t bytes) -> void* {
        void* p = base + off;
        off += (bytes + 255) & ~(size_t)255;
        return p;
    };
    unsigned short* wih_bf = (unsigned short*)carve((size_t)G4H * XW * 2);
    unsigned short* whh_bf = (unsigned short*)carve((size_t)G4H * HID * 2);
    unsigned short* wis_bf = (unsigned short*)carve((size_t)HID * CTXW * 2);
    unsigned short* wic_bf = (unsigned short*)carve((size_t)HID * CTXW * 2);
    unsigned short* wd1_bf = (unsigned short*)carve((size_t)HID * 2 * HID * 2);
    unsigned short* wd2_bf = (unsigned short*)carve((size_t)VOC * HID * 2);
    unsigned short* ctx_bf = (unsigned short*)carve((size_t)BB * CTXW * 2);
    unsigned short* xe     = (unsigned short*)carve((size_t)TT * BB * EMB * 2);
    float*          pre    = (float*)carve((size_t)TT * BB * G4H * 4);
    float*          ctxg   = (float*)carve((size_t)BB * G4H * 4);
    float*          gts    = (float*)carve((size_t)BB * G4H * 4);
    float*          cst    = (float*)carve((size_t)BB * HID * 4);
    unsigned short* hc     = (unsigned short*)carve((size_t)BB * 2 * HID * 2);
    unsigned short* hid    = (unsigned short*)carve((size_t)TT * BB * HID * 2);
    float*          bsum   = (float*)carve((size_t)G4H * 4);

    const int CT = 256;
    auto cvt = [&](const float* s, unsigned short* d, int n) {
        k_f32_to_bf16<<<(n + CT - 1) / CT, CT, 0, stream>>>(s, d, n);
    };

    // 1) fp32 -> bf16 conversions (weights stay L2-resident afterwards)
    cvt(W_ih,    wih_bf, G4H * XW);
    cvt(W_hh,    whh_bf, G4H * HID);
    cvt(W_initS, wis_bf, HID * CTXW);
    cvt(W_initC, wic_bf, HID * CTXW);
    cvt(W_d1,    wd1_bf, HID * 2 * HID);
    cvt(W_d2,    wd2_bf, VOC * HID);
    cvt(context, ctx_bf, BB * CTXW);
    k_bias_sum<<<(G4H + CT - 1) / CT, CT, 0, stream>>>(b_ih, b_hh, bsum, G4H);

    // 2) embedding gather+convert
    k_gather_emb<<<(TT * BB * EMB) / CT, CT, 0, stream>>>(seq, emb, xe);

    dim3 blk(128);

    // 3) init states (direct-load GEMMs; tanh epilogue)
    k_gemm_bf16<<<dim3(HID / 64, 1), blk, 0, stream>>>(
        ctx_bf, CTXW, wis_bf, CTXW, b_initS,
        nullptr, 0, 1, nullptr, 0, hc, 2 * HID, BB, HID, CTXW, 1);
    k_gemm_bf16<<<dim3(HID / 64, 1), blk, 0, stream>>>(
        ctx_bf, CTXW, wic_bf, CTXW, b_initC,
        nullptr, 0, 1, cst, HID, hc + HID, 2 * HID, BB, HID, CTXW, 1);

    // 4) loop-invariant gate part: ctxg = ctx @ W_ih[:, :CTX].T
    k_gemm_bf16<<<dim3(G4H / 64, 1), blk, 0, stream>>>(
        ctx_bf, CTXW, wih_bf, XW, nullptr,
        nullptr, 0, 1, ctxg, G4H, nullptr, 0, BB, G4H, CTXW, 0);

    // 5) token-parallel gate part (TDM-staged):
    //    pre[t*B+b] = xe @ W_ih[:, CTX:].T + ctxg[b] + (b_ih + b_hh)
    k_gemm_tdm<<<dim3(G4H / 64, (TT * BB) / 64), blk, 0, stream>>>(
        xe, EMB, wih_bf + CTXW, XW, bsum,
        ctxg, G4H, BB, pre, G4H, nullptr, 0, TT * BB, G4H, EMB, 0);

    // 6) sequential scan: only the true recurrence stays serial
    for (int t = 0; t < TT; ++t) {
        k_gemm_tdm<<<dim3(G4H / 64, 1), blk, 0, stream>>>(
            hc, 2 * HID, whh_bf, HID, nullptr,
            pre + (size_t)t * BB * G4H, G4H, BB,
            gts, G4H, nullptr, 0, BB, G4H, HID, 0);
        k_lstm_cell<<<(BB * HID) / CT, CT, 0, stream>>>(gts, cst, hc);
        k_gemm_tdm<<<dim3(HID / 64, 1), blk, 0, stream>>>(
            hc, 2 * HID, wd1_bf, 2 * HID, b_d1,
            nullptr, 0, 1, nullptr, 0,
            hid + (size_t)t * BB * HID, HID, BB, HID, 2 * HID, 1);
    }

    // 7) deferred vocab projection (bulk of FLOPs, TDM-staged)
    k_gemm_tdm<<<dim3(VOC / 64, (TT * BB) / 64), blk, 0, stream>>>(
        hid, HID, wd2_bf, HID, b_d2,
        nullptr, 0, 1, out, VOC, nullptr, 0, TT * BB, VOC, HID, 0);
}
